// EquivariantBlock_46755013984797
// MI455X (gfx1250) — compile-verified, matrix operand order checked by reading, not run on hardware
//
#include <hip/hip_runtime.h>
#include <cstdint>
#include <cstddef>

// Problem constants (match reference)
#define N_NODES 50000
#define N_EDGES 800000
#define DIM     128
#define SDIM    64
#define SH_DIM  16
#define EA_DIM  32
#define FC_DIM  128
#define NHEAD   8
#define AK      160   // 2*SD + EA
#define LN_EPS  1e-5f

typedef __attribute__((ext_vector_type(16))) _Float16 v16h;
typedef __attribute__((ext_vector_type(8)))  float    v8f;
typedef __attribute__((ext_vector_type(4)))  float    f32x4;
typedef __attribute__((ext_vector_type(4)))  unsigned u32x4;
typedef __attribute__((ext_vector_type(8)))  int      i32x8;
typedef __attribute__((ext_vector_type(4)))  int      i32x4;

// ---------------------------------------------------------------------------
// WMMA helper: D = A(16x32 f16) * B(32x16 f16) + C(16x16 f32)
// ---------------------------------------------------------------------------
__device__ __forceinline__ v8f wmma32(v16h a, v16h b, v8f c) {
  return __builtin_amdgcn_wmma_f32_16x16x32_f16(
      false, a, false, b, (short)0, c, false, false);
}

// ---------------------------------------------------------------------------
// Fragment-swizzled LDS layout: every 16x32 (A) or 32x16 (B) tile is stored as
// 512 halves where lane L's 16 fragment halves are contiguous at [L*16 .. L*16+15].
// Fragment loads become two ds_load_b128 per operand.
//
// A fragment (MxK 16x32): lane = r + 16*((k32>>3)&1), j = (k32&7) + ((k32>>4)&1)*8
// B fragment (KxN 32x16): lane = n + 16*((k32>>3)&1), j = (k32&7) + ((k32>>4)&1)*8
// ---------------------------------------------------------------------------
__device__ __forceinline__ int a_swz(int r, int c) {
  int k32 = c & 31;
  return (c >> 5) * 512 + (r + 16 * ((k32 >> 3) & 1)) * 16 + (k32 & 7) +
         ((k32 >> 4) & 1) * 8;
}

__device__ __forceinline__ v16h load_frag(const _Float16* buf, int tile, int lane) {
  return *(const v16h*)(buf + tile * 512 + lane * 16);
}

// Stage a row-major [K][Ncols] f32 weight into B-fragment-swizzled f16 LDS.
// Tile index = (k>>5) * (Ncols/16) + (c>>4).
__device__ __forceinline__ void stage_B_swz(_Float16* dst, const float* src, int K,
                                            int Ncols, int tid, int nthreads) {
  int NT = Ncols >> 4;
  for (int i = tid; i < K * Ncols; i += nthreads) {
    int k = i / Ncols, c = i % Ncols;
    int k32 = k & 31, n = c & 15;
    int lane = n + 16 * ((k32 >> 3) & 1);
    int j = (k32 & 7) + ((k32 >> 4) & 1) * 8;
    dst[((k >> 5) * NT + (c >> 4)) * 512 + lane * 16 + j] = (_Float16)src[i];
  }
}

__device__ __forceinline__ float fast_silu(float x) {
  return x * __builtin_amdgcn_rcpf(1.f + __expf(-x));
}

// Order-preserving float <-> uint key (for atomic max over floats)
__device__ __forceinline__ unsigned f2key(float f) {
  unsigned u = __float_as_uint(f);
  return (u & 0x80000000u) ? ~u : (u | 0x80000000u);
}
__device__ __forceinline__ float key2f(unsigned k) {
  unsigned u = (k & 0x80000000u) ? (k & 0x7FFFFFFFu) : ~k;
  return __uint_as_float(u);
}
#define NEG_INF_KEY 0x007FFFFFu  // f2key(-inf)

// ---------------------------------------------------------------------------
// K0: re-init scratch each launch (graph replays reuse ws)
// ---------------------------------------------------------------------------
__global__ __launch_bounds__(256) void k_init(float* agg, unsigned* hmax, float* hsum) {
  long idx = (long)blockIdx.x * 256 + threadIdx.x;
  if (idx < (long)N_NODES * DIM) agg[idx] = 0.f;
  if (idx < NHEAD) { hmax[idx] = NEG_INF_KEY; hsum[idx] = 0.f; }
}

// ---------------------------------------------------------------------------
// K1: message kernel. Per wave: 16-edge tile.
//   h1 = silu(ea@fc1+b1); h2 = silu(h1@fc2+b2); scale = h2@fc3+b3
//   msg = silu((x_src@W_node)*scale + edge_sh@W_sh)   -> f16 scratch
// ---------------------------------------------------------------------------
__global__ __launch_bounds__(128) void k_edge_message(
    const float* __restrict__ node_features, const float* __restrict__ edge_attr,
    const float* __restrict__ edge_sh,       const float* __restrict__ W_node,
    const float* __restrict__ fc1, const float* __restrict__ b1,
    const float* __restrict__ fc2, const float* __restrict__ b2,
    const float* __restrict__ fc3, const float* __restrict__ b3,
    const float* __restrict__ W_sh, const int* __restrict__ src_idx,
    _Float16* __restrict__ msg_out) {
  extern __shared__ __attribute__((aligned(64))) char smem_raw[];
  _Float16* sFc1 = (_Float16*)smem_raw;       // 32x128  -> 8 tiles
  _Float16* sFc2 = sFc1 + 32 * 128;           // 128x128 -> 32 tiles
  _Float16* sFc3 = sFc2 + 128 * 128;
  _Float16* sWn  = sFc3 + 128 * 128;
  _Float16* sWsh = sWn + 128 * 128;           // 32x128 (K padded 16->32)
  float*    sB1  = (float*)(sWsh + 32 * 128);
  float*    sB2  = sB1 + 128;
  float*    sB3  = sB2 + 128;
  _Float16* sStage = (_Float16*)(sB3 + 128);  // 4 waves * 2 bufs * 2048 halves

  const int tid = threadIdx.x;
  stage_B_swz(sFc1, fc1, 32, 128, tid, 128);
  stage_B_swz(sFc2, fc2, 128, 128, tid, 128);
  stage_B_swz(sFc3, fc3, 128, 128, tid, 128);
  stage_B_swz(sWn, W_node, 128, 128, tid, 128);
  for (int i = tid; i < 32 * 128; i += 128) {   // W_sh, K padded to 32
    int k = i >> 7, c = i & 127;
    int k32 = k & 31, n = c & 15;
    int lane = n + 16 * ((k32 >> 3) & 1);
    int j = (k32 & 7) + ((k32 >> 4) & 1) * 8;
    sWsh[(c >> 4) * 512 + lane * 16 + j] =
        (k < SH_DIM) ? (_Float16)W_sh[k * 128 + c] : (_Float16)0.f;
  }
  sB1[tid] = b1[tid]; sB2[tid] = b2[tid]; sB3[tid] = b3[tid];
  __syncthreads();

  const int wave = tid >> 5, lane = tid & 31;
  const long e0 = ((long)blockIdx.x * 4 + wave) * 16;
  _Float16* st0 = sStage + wave * 2 * 2048;  // 16x128 ping (A-swizzled, 4 tiles)
  _Float16* st1 = st0 + 2048;                // 16x128 pong
  const int r = lane & 15;
  const int k0 = (lane & 16) ? 8 : 0;
  const int half = (lane & 16) ? 8 : 0;
  const int n = lane & 15;

  // ---- A fragment of edge_attr (vectorized b128 global loads)
  v16h aEA;
  {
    const float* rp = edge_attr + (e0 + r) * EA_DIM + k0;
    f32x4 p0 = *(const f32x4*)(rp);
    f32x4 p1 = *(const f32x4*)(rp + 4);
    f32x4 p2 = *(const f32x4*)(rp + 16);
    f32x4 p3 = *(const f32x4*)(rp + 20);
#pragma unroll
    for (int j = 0; j < 4; ++j) {
      aEA[j] = (_Float16)p0[j]; aEA[j + 4] = (_Float16)p1[j];
      aEA[j + 8] = (_Float16)p2[j]; aEA[j + 12] = (_Float16)p3[j];
    }
  }

  // ---- h1 = silu(edge_attr @ fc1 + b1) -> st0 (A-swizzled)
  for (int t = 0; t < 8; ++t) {
    float bv = sB1[t * 16 + n];
    v8f c;
#pragma unroll
    for (int i = 0; i < 8; ++i) c[i] = bv;
    c = wmma32(aEA, load_frag(sFc1, t, lane), c);
#pragma unroll
    for (int i = 0; i < 8; ++i)
      st0[a_swz(i + half, t * 16 + n)] = (_Float16)fast_silu(c[i]);
  }

  // ---- h2 = silu(h1 @ fc2 + b2) -> st1
  for (int t = 0; t < 8; ++t) {
    float bv = sB2[t * 16 + n];
    v8f c;
#pragma unroll
    for (int i = 0; i < 8; ++i) c[i] = bv;
#pragma unroll
    for (int kk = 0; kk < 4; ++kk)
      c = wmma32(load_frag(st0, kk, lane), load_frag(sFc2, kk * 8 + t, lane), c);
#pragma unroll
    for (int i = 0; i < 8; ++i)
      st1[a_swz(i + half, t * 16 + n)] = (_Float16)fast_silu(c[i]);
  }

  // ---- stage gathered x_src tile into st0 (A-swizzled), 4-wide loads
  for (int i = lane * 4; i < 16 * 128; i += 32 * 4) {
    int rr = i >> 7, cc = i & 127;
    f32x4 v = *(const f32x4*)(node_features + (long)src_idx[e0 + rr] * DIM + cc);
#pragma unroll
    for (int j = 0; j < 4; ++j) st0[a_swz(rr, cc + j)] = (_Float16)v[j];
  }

  // ---- edge_sh A fragment, K padded 16 -> 32 with zeros
  v16h aSH;
  {
    const float* rp = edge_sh + (e0 + r) * SH_DIM + k0;
    f32x4 q0 = *(const f32x4*)(rp);
    f32x4 q1 = *(const f32x4*)(rp + 4);
#pragma unroll
    for (int j = 0; j < 4; ++j) {
      aSH[j] = (_Float16)q0[j]; aSH[j + 4] = (_Float16)q1[j];
      aSH[j + 8] = (_Float16)0.f; aSH[j + 12] = (_Float16)0.f;
    }
  }

  // ---- msg = silu((x@Wn)*scale + sh@Wsh)
  for (int t = 0; t < 8; ++t) {
    float bv = sB3[t * 16 + n];
    v8f sc;
#pragma unroll
    for (int i = 0; i < 8; ++i) sc[i] = bv;
    v8f xw;
#pragma unroll
    for (int i = 0; i < 8; ++i) xw[i] = 0.f;
#pragma unroll
    for (int kk = 0; kk < 4; ++kk) {
      sc = wmma32(load_frag(st1, kk, lane), load_frag(sFc3, kk * 8 + t, lane), sc);
      xw = wmma32(load_frag(st0, kk, lane), load_frag(sWn, kk * 8 + t, lane), xw);
    }
    v8f sh;
#pragma unroll
    for (int i = 0; i < 8; ++i) sh[i] = 0.f;
    sh = wmma32(aSH, load_frag(sWsh, t, lane), sh);
#pragma unroll
    for (int i = 0; i < 8; ++i) {
      float x = xw[i] * sc[i] + sh[i];
      msg_out[(e0 + i + half) * DIM + t * 16 + n] = (_Float16)fast_silu(x);
    }
  }
}

// ---------------------------------------------------------------------------
// K2: attention logits. alpha_in = [x_src[:64] | x_dst[:64] | edge_attr] (K=160)
// ---------------------------------------------------------------------------
__global__ __launch_bounds__(128) void k_edge_attn(
    const float* __restrict__ node_features, const float* __restrict__ edge_attr,
    const float* __restrict__ aW1, const float* __restrict__ ab1,
    const float* __restrict__ aW2, const float* __restrict__ ab2,
    const float* __restrict__ aW3, const float* __restrict__ ab3,
    const int* __restrict__ src_idx, const int* __restrict__ dst_idx,
    float* __restrict__ logits) {
  extern __shared__ __attribute__((aligned(64))) char smem_raw[];
  _Float16* sA1 = (_Float16*)smem_raw;     // 160x128 -> 40 tiles
  _Float16* sA2 = sA1 + AK * 128;          // 128x128 -> 32 tiles
  _Float16* sA3 = sA2 + 128 * 128;         // 128x16 (N padded 8->16) -> 4 tiles
  float*    sB1 = (float*)(sA3 + 128 * 16);
  float*    sB2 = sB1 + 128;
  float*    sB3 = sB2 + 128;               // 16 floats (padded)
  _Float16* sStage = (_Float16*)(sB3 + 16);

  const int tid = threadIdx.x;
  stage_B_swz(sA1, aW1, AK, 128, tid, 128);
  stage_B_swz(sA2, aW2, 128, 128, tid, 128);
  for (int i = tid; i < 128 * 16; i += 128) {   // aW3, N padded to 16
    int k = i >> 4, c = i & 15;
    int k32 = k & 31;
    int lane = c + 16 * ((k32 >> 3) & 1);
    int j = (k32 & 7) + ((k32 >> 4) & 1) * 8;
    sA3[(k >> 5) * 512 + lane * 16 + j] =
        (c < NHEAD) ? (_Float16)aW3[k * NHEAD + c] : (_Float16)0.f;
  }
  sB1[tid] = ab1[tid]; sB2[tid] = ab2[tid];
  if (tid < 16) sB3[tid] = (tid < NHEAD) ? ab3[tid] : 0.f;
  __syncthreads();

  const int wave = tid >> 5, lane = tid & 31;
  const long e0 = ((long)blockIdx.x * 4 + wave) * 16;
  _Float16* stIn = sStage + wave * (16 * AK + 16 * 128);  // 16x160 (5 tiles)
  _Float16* stA  = stIn + 16 * AK;                        // 16x128 (4 tiles)
  const int half = (lane & 16) ? 8 : 0;
  const int n = lane & 15;

  // stage alpha_in tile (A-swizzled), 4-wide loads
  for (int i = lane * 4; i < 16 * AK; i += 32 * 4) {
    int rr = i / AK, cc = i % AK;
    f32x4 v;
    if (cc < SDIM) {
      v = *(const f32x4*)(node_features + (long)src_idx[e0 + rr] * DIM + cc);
    } else if (cc < 2 * SDIM) {
      v = *(const f32x4*)(node_features + (long)dst_idx[e0 + rr] * DIM + (cc - SDIM));
    } else {
      v = *(const f32x4*)(edge_attr + (e0 + rr) * EA_DIM + (cc - 2 * SDIM));
    }
#pragma unroll
    for (int j = 0; j < 4; ++j) stIn[a_swz(rr, cc + j)] = (_Float16)v[j];
  }

  // a1 = silu(in @ aW1 + ab1) -> stA
  for (int t = 0; t < 8; ++t) {
    float bv = sB1[t * 16 + n];
    v8f c;
#pragma unroll
    for (int i = 0; i < 8; ++i) c[i] = bv;
#pragma unroll
    for (int kk = 0; kk < 5; ++kk)
      c = wmma32(load_frag(stIn, kk, lane), load_frag(sA1, kk * 8 + t, lane), c);
#pragma unroll
    for (int i = 0; i < 8; ++i)
      stA[a_swz(i + half, t * 16 + n)] = (_Float16)fast_silu(c[i]);
  }

  // a2 = silu(a1 @ aW2 + ab2) -> stIn tiles 0..3 (alpha_in no longer needed)
  for (int t = 0; t < 8; ++t) {
    float bv = sB2[t * 16 + n];
    v8f c;
#pragma unroll
    for (int i = 0; i < 8; ++i) c[i] = bv;
#pragma unroll
    for (int kk = 0; kk < 4; ++kk)
      c = wmma32(load_frag(stA, kk, lane), load_frag(sA2, kk * 8 + t, lane), c);
#pragma unroll
    for (int i = 0; i < 8; ++i)
      stIn[a_swz(i + half, t * 16 + n)] = (_Float16)fast_silu(c[i]);
  }

  // a3 = a2 @ aW3 + ab3 (single N-tile, N padded to 16)
  {
    float bv = sB3[n];
    v8f c;
#pragma unroll
    for (int i = 0; i < 8; ++i) c[i] = bv;
#pragma unroll
    for (int kk = 0; kk < 4; ++kk)
      c = wmma32(load_frag(stIn, kk, lane), load_frag(sA3, kk, lane), c);
    if (n < NHEAD) {
#pragma unroll
      for (int i = 0; i < 8; ++i)
        logits[(e0 + i + half) * NHEAD + n] = c[i];
    }
  }
}

// ---------------------------------------------------------------------------
// K3/K4: global softmax over the edge axis per head
// ---------------------------------------------------------------------------
__global__ __launch_bounds__(256) void k_head_max(const float* __restrict__ logits,
                                                  unsigned* __restrict__ hmax) {
  __shared__ unsigned red[NHEAD * 256];
  const int tid = threadIdx.x;
  float mx[NHEAD];
#pragma unroll
  for (int h = 0; h < NHEAD; ++h) mx[h] = -3.4e38f;
  for (long e = (long)blockIdx.x * 256 + tid; e < N_EDGES; e += (long)gridDim.x * 256) {
#pragma unroll
    for (int h = 0; h < NHEAD; ++h) mx[h] = fmaxf(mx[h], logits[e * NHEAD + h]);
  }
#pragma unroll
  for (int h = 0; h < NHEAD; ++h) red[h * 256 + tid] = f2key(mx[h]);
  __syncthreads();
  for (int s = 128; s > 0; s >>= 1) {
    if (tid < s) {
#pragma unroll
      for (int h = 0; h < NHEAD; ++h) {
        unsigned a = red[h * 256 + tid], b = red[h * 256 + tid + s];
        red[h * 256 + tid] = a > b ? a : b;
      }
    }
    __syncthreads();
  }
  if (tid < NHEAD) atomicMax(&hmax[tid], red[tid * 256]);
}

__global__ __launch_bounds__(256) void k_head_sum(const float* __restrict__ logits,
                                                  const unsigned* __restrict__ hmax,
                                                  float* __restrict__ hsum) {
  __shared__ float red[NHEAD * 256];
  const int tid = threadIdx.x;
  float mx[NHEAD], sm[NHEAD];
#pragma unroll
  for (int h = 0; h < NHEAD; ++h) { mx[h] = key2f(hmax[h]); sm[h] = 0.f; }
  for (long e = (long)blockIdx.x * 256 + tid; e < N_EDGES; e += (long)gridDim.x * 256) {
#pragma unroll
    for (int h = 0; h < NHEAD; ++h) sm[h] += __expf(logits[e * NHEAD + h] - mx[h]);
  }
#pragma unroll
  for (int h = 0; h < NHEAD; ++h) red[h * 256 + tid] = sm[h];
  __syncthreads();
  for (int s = 128; s > 0; s >>= 1) {
    if (tid < s) {
#pragma unroll
      for (int h = 0; h < NHEAD; ++h) red[h * 256 + tid] += red[h * 256 + tid + s];
    }
    __syncthreads();
  }
  if (tid < NHEAD) atomicAdd(&hsum[tid], red[tid * 256]);
}

__global__ __launch_bounds__(256) void k_alpha_mean(const float* __restrict__ logits,
                                                    const unsigned* __restrict__ hmax,
                                                    const float* __restrict__ hsum,
                                                    float* __restrict__ amean) {
  long e = (long)blockIdx.x * 256 + threadIdx.x;
  if (e >= N_EDGES) return;
  float acc = 0.f;
#pragma unroll
  for (int h = 0; h < NHEAD; ++h)
    acc += __expf(logits[e * NHEAD + h] - key2f(hmax[h])) *
           __builtin_amdgcn_rcpf(hsum[h]);
  amean[e] = acc * (1.f / NHEAD);
}

// ---------------------------------------------------------------------------
// K6: scatter-add  agg[dst] += msg * alpha_mean   (global f32 atomics)
// ---------------------------------------------------------------------------
__global__ __launch_bounds__(256) void k_scatter(const _Float16* __restrict__ msg,
                                                 const float* __restrict__ amean,
                                                 const int* __restrict__ dst_idx,
                                                 float* __restrict__ agg) {
  long idx = (long)blockIdx.x * 256 + threadIdx.x;
  if (idx >= (long)N_EDGES * DIM) return;
  long e = idx >> 7;
  int c = (int)(idx & 127);
  float v = (float)msg[e * DIM + c] * amean[e];
  atomicAdd(&agg[(long)dst_idx[e] * DIM + c], v);
}

// ---------------------------------------------------------------------------
// K7: out = layernorm(node_features + agg @ W_out)
//     W_out pulled into LDS by the Tensor Data Mover when available.
// ---------------------------------------------------------------------------
__global__ __launch_bounds__(128) void k_node_out(const float* __restrict__ node_features,
                                                  const float* __restrict__ agg,
                                                  const float* __restrict__ W_out,
                                                  float* __restrict__ out) {
  extern __shared__ __attribute__((aligned(64))) char smem_raw[];
  float*    sWf32  = (float*)smem_raw;            // 128x128 f32 (TDM destination)
  _Float16* sW     = (_Float16*)(sWf32 + 128 * 128);  // swizzled, 32 tiles
  _Float16* sStage = sW + 128 * 128;              // 4 waves * 16x128 (4 tiles)

#if __has_builtin(__builtin_amdgcn_tensor_load_to_lds) && __has_builtin(__builtin_amdgcn_s_wait_tensorcnt)
  if (threadIdx.x < 32) {
    unsigned long long ga = (unsigned long long)W_out;
    unsigned ldsa = (unsigned)(unsigned long long)(void*)sWf32;  // flat->LDS low 32 bits
    u32x4 g0;
    g0[0] = 1u;                                            // count=1, gather off
    g0[1] = ldsa;                                          // lds_addr
    g0[2] = (unsigned)(ga & 0xFFFFFFFFull);
    g0[3] = (unsigned)((ga >> 32) & 0x1FFFFFFull) | (2u << 30);  // addr hi | type=2
    i32x8 g1;
    g1[0] = (int)(2u << 16);          // data_size = 4 bytes
    g1[1] = (int)(128u << 16);        // tensor_dim0 = 128
    g1[2] = (int)(128u << 16);        // tensor_dim1 = 128
    g1[3] = (int)(128u << 16);        // tile_dim0 = 128
    g1[4] = 128;                      // tile_dim1 = 128, tile_dim2 = 0
    g1[5] = 128;                      // tensor_dim0_stride = 128
    g1[6] = 0;
    g1[7] = 0;
    i32x4 g2; g2[0] = 0; g2[1] = 0; g2[2] = 0; g2[3] = 0;
    i32x4 g3 = g2;
#if __clang_major__ >= 23
    i32x8 g4; for (int i = 0; i < 8; ++i) g4[i] = 0;
    __builtin_amdgcn_tensor_load_to_lds(g0, g1, g2, g3, g4, 0);
#else
    __builtin_amdgcn_tensor_load_to_lds(g0, g1, g2, g3, 0);
#endif
    __builtin_amdgcn_s_wait_tensorcnt(0);
  }
#else
  for (int i = threadIdx.x; i < 128 * 128; i += 128) sWf32[i] = W_out[i];
#endif
  __syncthreads();
  // convert + B-swizzle W_out into f16
  for (int i = threadIdx.x; i < 128 * 128; i += 128) {
    int k = i >> 7, c = i & 127;
    int k32 = k & 31, nn = c & 15;
    int lane = nn + 16 * ((k32 >> 3) & 1);
    int j = (k32 & 7) + ((k32 >> 4) & 1) * 8;
    sW[((k >> 5) * 8 + (c >> 4)) * 512 + lane * 16 + j] = (_Float16)sWf32[i];
  }
  __syncthreads();

  const int wave = threadIdx.x >> 5, lane = threadIdx.x & 31;
  const long n0 = ((long)blockIdx.x * 4 + wave) * 16;
  if (n0 >= N_NODES) return;  // fully OOB tile (no barriers below)
  _Float16* stAgg = sStage + wave * 2048;
  const int half = (lane & 16) ? 8 : 0;
  const int n = lane & 15;

  for (int i = lane * 4; i < 16 * 128; i += 32 * 4) {
    int rr = i >> 7, cc = i & 127;
    long nr = n0 + rr;
    if (nr < N_NODES) {
      f32x4 v = *(const f32x4*)(agg + nr * DIM + cc);
#pragma unroll
      for (int j = 0; j < 4; ++j) stAgg[a_swz(rr, cc + j)] = (_Float16)v[j];
    } else {
#pragma unroll
      for (int j = 0; j < 4; ++j) stAgg[a_swz(rr, cc + j)] = (_Float16)0.f;
    }
  }

  float ov[8][8], s1[8], s2[8];
#pragma unroll
  for (int i = 0; i < 8; ++i) { s1[i] = 0.f; s2[i] = 0.f; }
  for (int t = 0; t < 8; ++t) {
    v8f c;
#pragma unroll
    for (int i = 0; i < 8; ++i) c[i] = 0.f;
#pragma unroll
    for (int kk = 0; kk < 4; ++kk)
      c = wmma32(load_frag(stAgg, kk, lane), load_frag(sW, kk * 8 + t, lane), c);
#pragma unroll
    for (int i = 0; i < 8; ++i) {
      long nr = n0 + half + i;
      float nf = (nr < N_NODES) ? node_features[nr * DIM + t * 16 + n] : 0.f;
      float x = nf + c[i];
      ov[t][i] = x;
      s1[i] += x;
      s2[i] += x * x;
    }
  }
  // row reduction across the 16 lanes of each half-wave
#pragma unroll
  for (int i = 0; i < 8; ++i) {
#pragma unroll
    for (int off = 1; off < 16; off <<= 1) {
      s1[i] += __shfl_xor(s1[i], off, 16);
      s2[i] += __shfl_xor(s2[i], off, 16);
    }
  }
#pragma unroll
  for (int i = 0; i < 8; ++i) {
    long nr = n0 + half + i;
    if (nr >= N_NODES) continue;
    float mu = s1[i] * (1.f / DIM);
    float var = s2[i] * (1.f / DIM) - mu * mu;
    float inv = rsqrtf(var + LN_EPS);
#pragma unroll
    for (int t = 0; t < 8; ++t)
      out[nr * DIM + t * 16 + n] = (ov[t][i] - mu) * inv;
  }
}

// ---------------------------------------------------------------------------
// Host launcher
// ---------------------------------------------------------------------------
extern "C" void kernel_launch(void* const* d_in, const int* in_sizes, int n_in,
                              void* d_out, int out_size, void* d_ws, size_t ws_size,
                              hipStream_t stream) {
  (void)in_sizes; (void)n_in; (void)out_size; (void)ws_size;
  const float* node = (const float*)d_in[0];
  const float* ea   = (const float*)d_in[1];
  const float* esh  = (const float*)d_in[2];
  const float* Wn   = (const float*)d_in[3];
  const float* fc1  = (const float*)d_in[4];
  const float* b1   = (const float*)d_in[5];
  const float* fc2  = (const float*)d_in[6];
  const float* b2   = (const float*)d_in[7];
  const float* fc3  = (const float*)d_in[8];
  const float* b3   = (const float*)d_in[9];
  const float* Wsh  = (const float*)d_in[10];
  const float* aW1  = (const float*)d_in[11];
  const float* ab1  = (const float*)d_in[12];
  const float* aW2  = (const float*)d_in[13];
  const float* ab2  = (const float*)d_in[14];
  const float* aW3  = (const float*)d_in[15];
  const float* ab3  = (const float*)d_in[16];
  const float* Wout = (const float*)d_in[17];
  const int*   eidx = (const int*)d_in[18];
  const int*   src  = eidx;
  const int*   dst  = eidx + N_EDGES;

  // Workspace layout (~259 MB)
  char* ws = (char*)d_ws;
  size_t off = 0;
  _Float16* msg   = (_Float16*)(ws + off); off += (size_t)N_EDGES * DIM * sizeof(_Float16);
  float*   logits = (float*)(ws + off);    off += (size_t)N_EDGES * NHEAD * sizeof(float);
  float*   amean  = (float*)(ws + off);    off += (size_t)N_EDGES * sizeof(float);
  float*   agg    = (float*)(ws + off);    off += (size_t)N_NODES * DIM * sizeof(float);
  unsigned* hmax  = (unsigned*)(ws + off); off += 32;
  float*   hsum   = (float*)(ws + off);

  // K0: init (must run every call: the graph replays reuse ws)
  k_init<<<((long)N_NODES * DIM + 255) / 256, 256, 0, stream>>>(agg, hmax, hsum);

  // K1: messages
  size_t shm1 = (size_t)(32 + 128 + 128 + 128 + 32) * 128 * 2 + 3 * 128 * 4 +
                4 * 2 * 16 * 128 * 2;  // = 148,992 B
  k_edge_message<<<N_EDGES / 64, 128, shm1, stream>>>(
      node, ea, esh, Wn, fc1, b1, fc2, b2, fc3, b3, Wsh, src, msg);

  // K2: attention logits
  size_t shm2 = (size_t)(AK * 128 + 128 * 128 + 128 * 16) * 2 + (128 + 128 + 16) * 4 +
                4 * (16 * AK + 16 * 128) * 2;  // = 115,776 B
  k_edge_attn<<<N_EDGES / 64, 128, shm2, stream>>>(
      node, ea, aW1, ab1, aW2, ab2, aW3, ab3, src, dst, logits);

  // K3/K4: softmax over the edge axis
  k_head_max<<<512, 256, 0, stream>>>(logits, hmax);
  k_head_sum<<<512, 256, 0, stream>>>(logits, hmax, hsum);

  // K5: per-edge mean attention weight
  k_alpha_mean<<<(N_EDGES + 255) / 256, 256, 0, stream>>>(logits, hmax, hsum, amean);

  // K6: segment sum via global f32 atomics
  k_scatter<<<(unsigned)(((long)N_EDGES * DIM + 255) / 256), 256, 0, stream>>>(
      msg, amean, dst, agg);

  // K7: output projection + layernorm (TDM loads W_out into LDS)
  size_t shm3 = 128 * 128 * 4 + 128 * 128 * 2 + 4 * 16 * 128 * 2;  // = 114,688 B
  int node_tiles = (N_NODES + 15) / 16;       // 3125
  int node_blocks = (node_tiles + 3) / 4;     // 782
  k_node_out<<<node_blocks, 128, shm3, stream>>>(node, agg, Wout, (float*)d_out);
}
